// Bottleneck_MDTA_74929999446452
// MI455X (gfx1250) — matrix operand order repaired and checked by reading, not compiled, not run on hardware
//
#include <hip/hip_runtime.h>
#include <hip/hip_bf16.h>

typedef __attribute__((ext_vector_type(16))) _Float16 v16h;
typedef __attribute__((ext_vector_type(8)))  _Float16 v8h;
typedef __attribute__((ext_vector_type(4)))  _Float16 v4h;
typedef __attribute__((ext_vector_type(8)))  float    v8f;
typedef __attribute__((ext_vector_type(4)))  float    v4f;

// ---------------------------------------------------------------------------
// f32 -> f16 conversion, 4 elements / thread
// ---------------------------------------------------------------------------
__global__ __launch_bounds__(256) void k_f32_to_f16(const float* __restrict__ in,
                                                    _Float16* __restrict__ out, int n4) {
  int i = blockIdx.x * 256 + threadIdx.x;
  if (i < n4) {
    v4f v = ((const v4f*)in)[i];
    v4h h;
    h.x = (_Float16)v.x; h.y = (_Float16)v.y; h.z = (_Float16)v.z; h.w = (_Float16)v.w;
    ((v4h*)out)[i] = h;
  }
}

// ---------------------------------------------------------------------------
// Fold BN: bias_out[m] = b - mean*gamma/sqrt(var+eps)
// ---------------------------------------------------------------------------
__global__ __launch_bounds__(256) void k_fold_bias(const float* __restrict__ g,
                                                   const float* __restrict__ b,
                                                   const float* __restrict__ mean,
                                                   const float* __restrict__ var,
                                                   float* __restrict__ out, int n) {
  int i = blockIdx.x * 256 + threadIdx.x;
  if (i < n) {
    float s = g[i] * rsqrtf(var[i] + 1e-5f);
    out[i] = b[i] - mean[i] * s;
  }
}

// ---------------------------------------------------------------------------
// Pack weight matrix [M x K] f32 (optionally scaled by gamma/sqrt(var+eps))
// into WMMA 16x16x32 f16 A-fragment order:
//   packed[((kc*Mtiles + mt)*32 + lane)*16 + i]
//   m = mt*16 + (lane&15),  k = kc*32 + i + (lane&16 ? 16 : 0)
// ---------------------------------------------------------------------------
__global__ __launch_bounds__(256) void k_pack_w(const float* __restrict__ w,
                                                const float* __restrict__ g,
                                                const float* __restrict__ var,
                                                _Float16* __restrict__ out,
                                                int M, int K, int total) {
  int idx = blockIdx.x * 256 + threadIdx.x;
  if (idx >= total) return;
  int Mtiles = M >> 4;
  int i    = idx & 15;
  int lane = (idx >> 4) & 31;
  int rest = idx >> 9;
  int mt = rest % Mtiles;
  int kc = rest / Mtiles;
  int m = mt * 16 + (lane & 15);
  int k = kc * 32 + i + ((lane & 16) ? 16 : 0);
  float s = 1.0f;
  if (g) s = g[m] * rsqrtf(var[m] + 1e-5f);
  out[idx] = (_Float16)(w[(size_t)m * K + k] * s);
}

// ---------------------------------------------------------------------------
// Generic WMMA implicit-GEMM: out[M x 4096] = Apacked[M x K] * B
//   CONV3: B = im2col(3x3, pad 1) of f16 NCHW planes, K = Cin*9 (gather)
//   else : 1x1, K = Cin; staging is fully coalesced v16h loads along N
// Block tile 128(M) x 128(N), 8 waves, each wave 2x4 WMMA tiles.
// A fragments direct from global (pre-swizzled). B double-buffered in LDS in
// fragment order: one barrier per K-chunk; next chunk's global fetch is issued
// before the barrier so it overlaps the current chunk's WMMAs.
// MODE: 0 = f16 store, 1 = +bias,SiLU,f16, 2 = f32 + residual.
// ---------------------------------------------------------------------------
template<int MODE, bool CONV3>
__global__ __launch_bounds__(256) void k_gemm_wmma(
    const _Float16* __restrict__ Ap, const _Float16* __restrict__ in,
    void* __restrict__ outv, const float* __restrict__ bias,
    const float* __restrict__ residual,
    int M, int K, int zshift,
    long a_zb, long a_zh, long in_zb, long in_zh, long out_zb, long out_zh)
{
  __shared__ _Float16 ldsB[2][4096];        // 2 x (32K x 128N) chunk, fragment order

  const int t    = threadIdx.x;
  const int lane = t & 31;
  const int wid  = t >> 5;
  const int wm   = wid & 3;                 // wave m position (0..3) -> 32 rows
  const int wn   = wid >> 2;                // wave n position (0..1) -> 64 cols

  const int z  = blockIdx.z;
  const int zb = z >> zshift;
  const int zh = z & ((1 << zshift) - 1);

  const _Float16* __restrict__ inB = in + (size_t)zb * in_zb + (size_t)zh * in_zh;
  const _Float16* __restrict__ ApB = Ap + (size_t)zb * a_zb  + (size_t)zh * a_zh;

  const int n0     = blockIdx.x * 128;
  const int mtile0 = blockIdx.y * 8;
  const int Mtiles = M >> 4;
  const int KC     = K >> 5;

  v8f zero = {};
  v8f acc[2][4];
#pragma unroll
  for (int a_ = 0; a_ < 2; ++a_)
#pragma unroll
    for (int b_ = 0; b_ < 4; ++b_) acc[a_][b_] = zero;

  // --- staging thread roles ---
  // CONV3 gather role: thread t owns LDS fragment slot t = (nt, lane)
  const int g_nl    = lane & 15;
  const int g_khalf = (lane & 16) ? 16 : 0;
  const int g_nf    = n0 + (t >> 5) * 16 + g_nl;
  const int g_py    = g_nf >> 6;
  const int g_px    = g_nf & 63;
  // 1x1 coalesced role: thread t loads K-row (t>>3), N-segment (t&7)
  const int c_kl   = t >> 3;                // 0..31
  const int c_nseg = t & 7;                 // 0..7
  const int c_khi  = c_kl & 16;             // lane-half offset in fragment order
  const int c_ki   = c_kl & 15;             // i within fragment

  auto fetch = [&](int kc, v16h& bv) {
    if constexpr (CONV3) {
#pragma unroll
      for (int i = 0; i < 16; ++i) {
        int k  = kc * 32 + g_khalf + i;
        int ci = k / 9;
        int rs = k - ci * 9;
        int r  = rs / 3;
        int s  = rs - r * 3;
        int iy = g_py + r - 1;
        int ix = g_px + s - 1;
        bv[i] = (iy >= 0 && iy < 64 && ix >= 0 && ix < 64)
                  ? inB[((size_t)ci << 12) + (iy << 6) + ix]
                  : (_Float16)0.0f;
      }
    } else {
      bv = *(const v16h*)(inB + (((size_t)(kc * 32 + c_kl)) << 12) + n0 + c_nseg * 16);
    }
  };
  auto stash = [&](int buf, v16h bv) {
    if constexpr (CONV3) {
      *(v16h*)&ldsB[buf][t << 4] = bv;
    } else {
#pragma unroll
      for (int j = 0; j < 16; ++j)
        ldsB[buf][(((c_nseg * 32) + j + c_khi) << 4) + c_ki] = bv[j];
    }
  };

  v16h bv;
  fetch(0, bv);
  for (int kc = 0; kc < KC; ++kc) {
    const int buf = kc & 1;
    stash(buf, bv);
    if (kc + 1 < KC) fetch(kc + 1, bv);     // in-flight during this chunk's WMMAs
    __syncthreads();

    v16h bfrag[4];
#pragma unroll
    for (int tn = 0; tn < 4; ++tn)
      bfrag[tn] = *(const v16h*)&ldsB[buf][(((wn * 4 + tn) * 32 + lane) << 4)];

#pragma unroll
    for (int tm = 0; tm < 2; ++tm) {
      int mtile = mtile0 + wm * 2 + tm;
      if (mtile < Mtiles) {
        v16h a = *(const v16h*)(ApB + (((size_t)kc * Mtiles + mtile) * 32 + lane) * 16);
#pragma unroll
        for (int tn = 0; tn < 4; ++tn) {
          acc[tm][tn] = __builtin_amdgcn_wmma_f32_16x16x32_f16(
              false, a, false, bfrag[tn], (short)0, acc[tm][tn], false, false);
        }
      }
    }
    // single barrier per iteration: separates this chunk's LDS reads from the
    // write of the same buffer two iterations later
  }

  // ---- epilogue ----
  float*    outF = (float*)outv;
  _Float16* outH = (_Float16*)outv;
  const size_t obase = (size_t)zb * out_zb + (size_t)zh * out_zh;
  const int rowhi = (lane & 16) ? 8 : 0;
  const int cl    = lane & 15;

#pragma unroll
  for (int tm = 0; tm < 2; ++tm) {
    int mtile = mtile0 + wm * 2 + tm;
    if (mtile >= Mtiles) continue;
#pragma unroll
    for (int tn = 0; tn < 4; ++tn) {
      int nn = n0 + (wn * 4 + tn) * 16 + cl;
#pragma unroll
      for (int r = 0; r < 8; ++r) {
        int m = mtile * 16 + rowhi + r;
        float v = acc[tm][tn][r];
        size_t idx = obase + ((size_t)m << 12) + nn;
        if (MODE == 1) {
          v += bias[m];
          v = v / (1.0f + __expf(-v));      // SiLU
          outH[idx] = (_Float16)v;
        } else if (MODE == 2) {
          outF[idx] = v + residual[idx];    // residual add (x + att)
        } else {
          outH[idx] = (_Float16)v;
        }
      }
    }
  }
}

// ---------------------------------------------------------------------------
// Depthwise 3x3 (pad 1) on 768 channels, f16 in/out; fused per-channel sum(y^2)
// ---------------------------------------------------------------------------
__global__ __launch_bounds__(256) void k_dwconv3(const _Float16* __restrict__ in,
                                                 const float* __restrict__ w,
                                                 _Float16* __restrict__ out,
                                                 float* __restrict__ sumsq) {
  __shared__ float red[256];
  int z  = blockIdx.x;                 // b*768 + ch
  int ch = z % 768;
  const _Float16* __restrict__ p = in + ((size_t)z << 12);
  _Float16* __restrict__ o = out + ((size_t)z << 12);
  float wk[9];
#pragma unroll
  for (int j = 0; j < 9; ++j) wk[j] = w[ch * 9 + j];

  float ss = 0.0f;
  for (int j = 0; j < 16; ++j) {
    int pix = threadIdx.x + j * 256;
    int y = pix >> 6, x = pix & 63;
    float acc = 0.0f;
#pragma unroll
    for (int r = 0; r < 3; ++r) {
#pragma unroll
      for (int s = 0; s < 3; ++s) {
        int iy = y + r - 1, ix = x + s - 1;
        if (iy >= 0 && iy < 64 && ix >= 0 && ix < 64)
          acc += wk[r * 3 + s] * (float)p[(iy << 6) + ix];
      }
    }
    o[pix] = (_Float16)acc;
    ss += acc * acc;
  }
  red[threadIdx.x] = ss;
  __syncthreads();
  for (int s = 128; s > 0; s >>= 1) {
    if (threadIdx.x < s) red[threadIdx.x] += red[threadIdx.x + s];
    __syncthreads();
  }
  if (threadIdx.x == 0) sumsq[z] = red[0];
}

// ---------------------------------------------------------------------------
// L2-normalize q,k planes in place: x *= 1/max(sqrt(sumsq), 1e-12)
// ---------------------------------------------------------------------------
__global__ __launch_bounds__(256) void k_l2norm(_Float16* __restrict__ qk,
                                                const float* __restrict__ sumsq) {
  int z  = blockIdx.x;                 // b*512 + ch,  ch in [0,512)
  int b  = z >> 9;
  int ch = z & 511;
  float f = 1.0f / fmaxf(sqrtf(sumsq[b * 768 + ch]), 1e-12f);
  _Float16* __restrict__ p = qk + (((size_t)b * 768 + ch) << 12);
  for (int j = 0; j < 16; ++j) {
    int pix = threadIdx.x + j * 256;
    p[pix] = (_Float16)((float)p[pix] * f);
  }
}

// ---------------------------------------------------------------------------
// attn = softmax(q_n @ k_n^T * temperature) per (b,head).
// M=N=64, K=4096. 4 waves; wave w owns m-tile w, 4 n-tiles. Ping-pong 2x
// unrolled register pipeline (no inter-stage copies): stage B's fragment loads
// are in flight while stage A's WMMAs execute, and vice versa.
// Softmax in LDS; result written pre-swizzled into WMMA A-fragment order.
// ---------------------------------------------------------------------------
__global__ __launch_bounds__(128) void k_attn_qk(const _Float16* __restrict__ dwqkv,
                                                 const float* __restrict__ temperature,
                                                 _Float16* __restrict__ apacked) {
  __shared__ float attnL[64 * 64];
  int z = blockIdx.x;                  // b*4 + h
  int b = z >> 2, h = z & 3;
  const _Float16* __restrict__ qb = dwqkv + (((size_t)b * 768 + h * 64) << 12);
  const _Float16* __restrict__ kb = dwqkv + (((size_t)b * 768 + 256 + h * 64) << 12);

  int t = threadIdx.x, lane = t & 31, wid = t >> 5;
  int mrow = lane & 15;

  v8f zero = {};
  v8f acc[4];
#pragma unroll
  for (int tn = 0; tn < 4; ++tn) acc[tn] = zero;

  const _Float16* __restrict__ qrow = qb + ((size_t)(wid * 16 + mrow) << 12);
  const int aoff = (lane & 16) ? 16 : 0;
  const int boff = (lane & 16) ? 16 : 0;

  auto ldfrag = [&](int kc, v16h& a, v16h* bf) {
    int k0 = kc * 32;
    v8h alo = *(const v8h*)(qrow + k0 + aoff);
    v8h ahi = *(const v8h*)(qrow + k0 + aoff + 8);
    a = __builtin_shufflevector(alo, ahi, 0, 1, 2, 3, 4, 5, 6, 7,
                                8, 9, 10, 11, 12, 13, 14, 15);
#pragma unroll
    for (int tn = 0; tn < 4; ++tn) {
      const _Float16* krow = kb + ((size_t)(tn * 16 + mrow) << 12) + k0 + boff;
      v8h b0 = *(const v8h*)krow;
      v8h b1 = *(const v8h*)(krow + 8);
      bf[tn] = __builtin_shufflevector(b0, b1, 0, 1, 2, 3, 4, 5, 6, 7,
                                       8, 9, 10, 11, 12, 13, 14, 15);
    }
  };
  auto domma = [&](const v16h& a, const v16h* bf) {
#pragma unroll
    for (int tn = 0; tn < 4; ++tn)
      acc[tn] = __builtin_amdgcn_wmma_f32_16x16x32_f16(
          false, a, false, bf[tn], (short)0, acc[tn], false, false);
  };

  v16h a0, a1;
  v16h b0[4], b1[4];
  ldfrag(0, a0, b0);
  for (int kc = 0; kc < 128; kc += 2) {     // K/32 = 128 chunks, exact 2x unroll
    ldfrag(kc + 1, a1, b1);                 // overlaps stage-A WMMAs
    domma(a0, b0);
    if (kc + 2 < 128) ldfrag(kc + 2, a0, b0);  // overlaps stage-B WMMAs
    domma(a1, b1);
  }

  int rowhi = (lane & 16) ? 8 : 0;
#pragma unroll
  for (int tn = 0; tn < 4; ++tn)
#pragma unroll
    for (int r = 0; r < 8; ++r)
      attnL[(wid * 16 + rowhi + r) * 64 + tn * 16 + mrow] = acc[tn][r];
  __syncthreads();

  if (t < 64) {
    float tp = temperature[h];
    float* row = &attnL[t * 64];
    float mx = -1e30f;
    for (int d = 0; d < 64; ++d) { float v = row[d] * tp; row[d] = v; mx = fmaxf(mx, v); }
    float s = 0.0f;
    for (int d = 0; d < 64; ++d) { float e = __expf(row[d] - mx); row[d] = e; s += e; }
    float inv = 1.0f / s;
    _Float16* ap = apacked + ((size_t)z << 12);
    int mt2 = t >> 4;
    int mr  = t & 15;
    for (int d = 0; d < 64; ++d) {
      float pv = row[d] * inv;
      int kc2 = d >> 5;
      int kl  = d & 31;
      int lhi = (kl >> 4) & 1;
      int i   = kl & 15;
      int ln  = mr + (lhi ? 16 : 0);
      ap[(((kc2 * 4 + mt2) * 32) + ln) * 16 + i] = (_Float16)pv;
    }
  }
}

// ---------------------------------------------------------------------------
extern "C" void kernel_launch(void* const* d_in, const int* in_sizes, int n_in,
                              void* d_out, int out_size, void* d_ws, size_t ws_size,
                              hipStream_t stream) {
  (void)in_sizes; (void)n_in; (void)out_size; (void)ws_size;

  const float* x       = (const float*)d_in[0];
  const float* cv1_w   = (const float*)d_in[1];
  const float* cv1_g   = (const float*)d_in[2];
  const float* cv1_b   = (const float*)d_in[3];
  const float* cv1_m   = (const float*)d_in[4];
  const float* cv1_v   = (const float*)d_in[5];
  const float* cv2_w   = (const float*)d_in[6];
  const float* cv2_g   = (const float*)d_in[7];
  const float* cv2_b   = (const float*)d_in[8];
  const float* cv2_m   = (const float*)d_in[9];
  const float* cv2_v   = (const float*)d_in[10];
  const float* qkv_w   = (const float*)d_in[11];
  const float* dw_w    = (const float*)d_in[12];
  const float* proj_w  = (const float*)d_in[13];
  const float* temper  = (const float*)d_in[14];

  const int B = 16, C = 256, HW = 4096;
  char* ws = (char*)d_ws;
  size_t off = 0;
  auto alloc = [&](size_t bytes) {
    size_t o = off;
    off += (bytes + 255) & ~(size_t)255;
    return o;
  };

  _Float16* xh     = (_Float16*)(ws + alloc((size_t)B * C * HW * 2));       // 33.5 MB
  _Float16* y1     = (_Float16*)(ws + alloc((size_t)B * 128 * HW * 2));     // 16.8 MB
  _Float16* y2     = (_Float16*)(ws + alloc((size_t)B * C * HW * 2));       // 33.5 MB
  _Float16* qkvb   = (_Float16*)(ws + alloc((size_t)B * 768 * HW * 2));     // 100 MB
  _Float16* dwqkv  = (_Float16*)(ws + alloc((size_t)B * 768 * HW * 2));     // 100 MB
  _Float16* w1p    = (_Float16*)(ws + alloc((size_t)128 * 2304 * 2));
  _Float16* w2p    = (_Float16*)(ws + alloc((size_t)256 * 1152 * 2));
  _Float16* wqkvp  = (_Float16*)(ws + alloc((size_t)768 * 256 * 2));
  _Float16* wprojp = (_Float16*)(ws + alloc((size_t)256 * 256 * 2));
  float*    bias1  = (float*)(ws + alloc(128 * 4));
  float*    bias2  = (float*)(ws + alloc(256 * 4));
  float*    sumsq  = (float*)(ws + alloc((size_t)B * 768 * 4));
  _Float16* attnP  = (_Float16*)(ws + alloc((size_t)B * 4 * 4096 * 2));
  _Float16* attn_out = xh;   // reuse: x_f16 is dead after conv1

  // 1. x -> f16
  {
    int n4 = B * C * HW / 4;
    k_f32_to_f16<<<dim3((n4 + 255) / 256), 256, 0, stream>>>(x, xh, n4);
  }
  // 2. fold BN biases
  k_fold_bias<<<dim3(1), 256, 0, stream>>>(cv1_g, cv1_b, cv1_m, cv1_v, bias1, 128);
  k_fold_bias<<<dim3(1), 256, 0, stream>>>(cv2_g, cv2_b, cv2_m, cv2_v, bias2, 256);
  // 3. pack weights into WMMA A-fragment order (BN scale folded for convs)
  {
    int t1 = (128 / 16) * (2304 / 32) * 512;
    k_pack_w<<<dim3((t1 + 255) / 256), 256, 0, stream>>>(cv1_w, cv1_g, cv1_v, w1p, 128, 2304, t1);
    int t2 = (256 / 16) * (1152 / 32) * 512;
    k_pack_w<<<dim3((t2 + 255) / 256), 256, 0, stream>>>(cv2_w, cv2_g, cv2_v, w2p, 256, 1152, t2);
    int t3 = (768 / 16) * (256 / 32) * 512;
    k_pack_w<<<dim3((t3 + 255) / 256), 256, 0, stream>>>(qkv_w, nullptr, nullptr, wqkvp, 768, 256, t3);
    int t4 = (256 / 16) * (256 / 32) * 512;
    k_pack_w<<<dim3((t4 + 255) / 256), 256, 0, stream>>>(proj_w, nullptr, nullptr, wprojp, 256, 256, t4);
  }
  // 4. cv1: 3x3 conv(256->128) + BN + SiLU
  k_gemm_wmma<1, true><<<dim3(32, 1, B), 256, 0, stream>>>(
      w1p, xh, y1, bias1, nullptr, 128, 2304, 0,
      0L, 0L, (long)256 * HW, 0L, (long)128 * HW, 0L);
  // 5. cv2: 3x3 conv(128->256) + BN + SiLU
  k_gemm_wmma<1, true><<<dim3(32, 2, B), 256, 0, stream>>>(
      w2p, y1, y2, bias2, nullptr, 256, 1152, 0,
      0L, 0L, (long)128 * HW, 0L, (long)256 * HW, 0L);
  // 6. qkv 1x1 conv(256->768)
  k_gemm_wmma<0, false><<<dim3(32, 6, B), 256, 0, stream>>>(
      wqkvp, y2, qkvb, nullptr, nullptr, 768, 256, 0,
      0L, 0L, (long)256 * HW, 0L, (long)768 * HW, 0L);
  // 7. depthwise 3x3 (+ per-channel sum of squares)
  k_dwconv3<<<dim3(B * 768), 256, 0, stream>>>(qkvb, dw_w, dwqkv, sumsq);
  // 8. L2-normalize q and k planes
  k_l2norm<<<dim3(B * 512), 256, 0, stream>>>(dwqkv, sumsq);
  // 9. attn = softmax(q @ k^T * temp), written pre-swizzled as packed A
  k_attn_qk<<<dim3(B * 4), 128, 0, stream>>>(dwqkv, temper, attnP);
  // 10. out = attn @ v   (per (b,head): M=64, K=64, N=4096)
  k_gemm_wmma<0, false><<<dim3(32, 1, B * 4), 256, 0, stream>>>(
      attnP, dwqkv + (size_t)512 * HW, attn_out, nullptr, nullptr, 64, 64, 2,
      (long)4 * 4096, (long)4096,
      (long)768 * HW, (long)64 * HW,
      (long)256 * HW, (long)64 * HW);
  // 11. proj 1x1 conv(256->256) + residual x  -> d_out (f32)
  k_gemm_wmma<2, false><<<dim3(32, 2, B), 256, 0, stream>>>(
      wprojp, attn_out, d_out, nullptr, x, 256, 256, 0,
      0L, 0L, (long)256 * HW, 0L, (long)256 * HW, 0L);
}
